// MultiHeadedAttention_10170482557739
// MI455X (gfx1250) — matrix operand order repaired and checked by reading, not compiled, not run on hardware
//
#include <hip/hip_runtime.h>

// ---------------------------------------------------------------------------
// MHA forward for B=4, S=2048, D=1024, H=16, DK=64 on gfx1250 (MI455X).
//   - one-shot f32->bf16 convert pass (inputs + weights)
//   - projection GEMMs: A tile staged by TDM tensor_load_to_lds (TENSORcnt),
//     B tile by global_load_async_to_lds_b128 (ASYNCcnt), double-buffered,
//     pure ds_load_b128 + v_wmma_f32_16x16x32_bf16 inner loop
//   - flash attention with online softmax, causal mask applied analytically
// ---------------------------------------------------------------------------

#define Bn 4
#define Sn 2048
#define Dn 1024
#define Hn 16
#define DKn 64
#define Mn (Bn * Sn)      // 8192 rows for the projection GEMMs
#define KST 72            // padded LDS row stride for 64-wide K step (144B)
#define LDST 40           // padded LDS row stride for attention P tile (80B)

typedef unsigned short u16;
typedef __attribute__((ext_vector_type(16))) __bf16 v16bf;
typedef __attribute__((ext_vector_type(8)))  float  v8f;
typedef __attribute__((ext_vector_type(4)))  unsigned int u32x4;
typedef __attribute__((ext_vector_type(4)))  int i32x4;
typedef __attribute__((ext_vector_type(8)))  int i32x8;

union Frag { u32x4 q[2]; v16bf v; };

// ---- f32 -> bf16 conversion helpers (RNE) ----
__device__ inline u16 f2bf(float f) {
  unsigned u = __float_as_uint(f);
  u += 0x7FFFu + ((u >> 16) & 1u);
  return (u16)(u >> 16);
}
__device__ inline unsigned pk2(float lo, float hi) {
#if __has_builtin(__builtin_amdgcn_cvt_pk_bf16_f32)
  typedef __attribute__((ext_vector_type(2))) __bf16 v2bf;
  union { v2bf v; unsigned u; } x;
  x.v = __builtin_amdgcn_cvt_pk_bf16_f32(lo, hi);
  return x.u;
#else
  return (unsigned)f2bf(lo) | ((unsigned)f2bf(hi) << 16);
#endif
}

__device__ inline v8f wmma_bf16(v16bf a, v16bf b, v8f c) {
  // (neg_a, A, neg_b, B, c_mod, C, reuse_a, reuse_b)
  return __builtin_amdgcn_wmma_f32_16x16x32_bf16(false, a, false, b, (short)0, c,
                                                 false, false);
}

// A-matrix 16x32 bf16 fragment (ISA 7.12.2): lane L holds row (L&15);
// lanes 0-15 get K = {0..7, 16..23}, lanes 16-31 get K = {8..15, 24..31}.
__device__ inline v16bf load_a_frag(const u16* p, int stride, int lane) {
  const int row = lane & 15, kh = lane >> 4;
  const u16* b = p + (size_t)row * stride + kh * 8;
  Frag f;
  f.q[0] = *(const u32x4*)(b);        // K = kh*8 + 0..7
  f.q[1] = *(const u32x4*)(b + 16);   // K = kh*8 + 16..23
  return f.v;
}

// B-matrix 32x16 bf16 fragment loaded from B^T stored row-major (N x K):
// lane L holds column N=(L&15); lanes 0-15: K={0..7,16..23}, lanes 16-31: K={8..15,24..31}.
__device__ inline v16bf load_bt_frag(const u16* p, int stride, int lane) {
  const int n = lane & 15, kh = lane >> 4;
  const u16* b = p + (size_t)n * stride + kh * 8;
  Frag f;
  f.q[0] = *(const u32x4*)(b);        // K = kh*8 + 0..7
  f.q[1] = *(const u32x4*)(b + 16);   // K = kh*8 + 16..23
  return f.v;
}

// ---- CDNA5 async global->LDS staging (ASYNCcnt-tracked) ----
// ldsoff: wave-relative LDS byte offset (low 32 bits of generic pointer).
__device__ inline void async_ld_b128(unsigned ldsoff, const u16* g) {
  asm volatile("global_load_async_to_lds_b128 %0, %1, off"
               :: "v"(ldsoff), "v"(g) : "memory");
}
__device__ inline void stage64B(unsigned ldsoff, const u16* g) {
  async_ld_b128(ldsoff,      g);
  async_ld_b128(ldsoff + 16, g + 8);
  async_ld_b128(ldsoff + 32, g + 16);
  async_ld_b128(ldsoff + 48, g + 24);
}
#define WAIT_ASYNC(n) asm volatile("s_wait_asynccnt %0" :: "n"(n) : "memory")

__device__ inline unsigned lds_off(const void* p) {
  return (unsigned)(size_t)p;   // generic LDS pointer: low 32 bits = LDS offset
}

// ---- CDNA5 Tensor Data Mover: 2D 64x64 bf16 tile, global -> LDS ----
// D# per ISA §8.3/§8.4; <=2D so the 2-SGPR-group form is used.
// LDS row stride = 128B data + 16B pad (pad_interval=4 -> 32 DW, pad_amount=3
// -> 4 DW) == KST(72) bf16 elements, matching the WMMA fragment loaders.
__device__ inline void tdm_load_tile64x64(unsigned lds_byte, const u16* gtile,
                                          int tensor_rows) {
  const unsigned long long ga = (unsigned long long)(size_t)gtile;
  i32x4 g0;
  g0[0] = 1;                                          // count=1 (valid, user)
  g0[1] = (int)lds_byte;                              // lds_addr
  g0[2] = (int)(unsigned)(ga & 0xFFFFFFFFull);        // global_addr[31:0]
  g0[3] = (int)((unsigned)((ga >> 32) & 0x01FFFFFFu)  // global_addr[56:32]
                | 0x80000000u);                       // type=2 ("image")
  i32x8 g1;
  g1[0] = 0x07110000;            // data_size=2B, pad_en, interval=32DW, pad=4DW
  g1[1] = (int)(Dn << 16);       // [63:48] tensor_dim0 = 1024 (lo16)
  g1[2] = (int)((tensor_rows & 0xFFFF) << 16);  // [111:80] tensor_dim1 (lo16)
  g1[3] = (64 << 16);            // [127:112] tile_dim0 = 64
  g1[4] = 64;                    // [143:128] tile_dim1 = 64, tile_dim2 = 0
  g1[5] = Dn;                    // [191:160] tensor_dim0_stride = 1024 (lo32)
  g1[6] = 0;                     // stride hi bits / dim1_stride lo = 0
  g1[7] = 0;
  asm volatile("tensor_load_to_lds %0, %1" :: "s"(g0), "s"(g1) : "memory");
}

// lane-half (16-wide) reductions matching C-matrix N striping
__device__ inline float half16_max(float x) {
  x = fmaxf(x, __shfl_xor(x, 1, 32));
  x = fmaxf(x, __shfl_xor(x, 2, 32));
  x = fmaxf(x, __shfl_xor(x, 4, 32));
  x = fmaxf(x, __shfl_xor(x, 8, 32));
  return x;
}
__device__ inline float half16_sum(float x) {
  x += __shfl_xor(x, 1, 32);
  x += __shfl_xor(x, 2, 32);
  x += __shfl_xor(x, 4, 32);
  x += __shfl_xor(x, 8, 32);
  return x;
}

// ---------------------------------------------------------------------------
// f32 -> bf16 bulk convert (n must be a multiple of 8)
// ---------------------------------------------------------------------------
__global__ __launch_bounds__(256)
void cvt_bf16_kernel(const float* __restrict__ src, u16* __restrict__ dst, int n) {
  const int i = (blockIdx.x * 256 + threadIdx.x) * 8;
  if (i >= n) return;
  const float4* s = (const float4*)(src + i);
  float4 a = s[0], b = s[1];
  u32x4 o;
  o.x = pk2(a.x, a.y);
  o.y = pk2(a.z, a.w);
  o.z = pk2(b.x, b.y);
  o.w = pk2(b.z, b.w);
  *(u32x4*)(dst + i) = o;
}

// ---------------------------------------------------------------------------
// Projection GEMM: Out[m,n] = sum_k A[m,k] * W[n,k] + bias[n], all bf16 in.
// MODE 0: bf16 out, head-split   [B,H,S,DK]
// MODE 1: bf16 out, head-split-T [B,H,DK,S]   (for V)
// MODE 2: f32 out, row-major [Mn, Dn]         (final output)
// Block: 128 threads (4 waves). Tile: 64(M) x 64(N), K-steps of 64.
// A tile staged by TDM (wave 0), B tile by async-to-LDS; double-buffered.
// ---------------------------------------------------------------------------
template <int MODE>
__global__ __launch_bounds__(128)
void mha_proj_kernel(const u16* __restrict__ A, const u16* __restrict__ Wb,
                     const float* __restrict__ bias, void* __restrict__ Out) {
  __shared__ u16 Abuf[2][64 * KST];
  __shared__ u16 Bbuf[2][64 * KST];

  const int tid  = threadIdx.x;
  const int lane = tid & 31;
  const int wv   = tid >> 5;          // wave id 0..3 -> M sub-tile
  const int m0   = blockIdx.x * 64;
  const int n0   = blockIdx.y * 64;
  const int row  = tid >> 1;          // 0..63: B staging row
  const int half = tid & 1;           // 32-element half of the 64-wide K step

  const u16* Arow0 = A + (size_t)m0 * Dn;                       // TDM source
  const u16* bg = Wb + (size_t)(n0 + row) * Dn + half * 32;     // async source
  const unsigned abase[2] = { lds_off(&Abuf[0][0]), lds_off(&Abuf[1][0]) };
  const unsigned boff[2]  = { lds_off(&Bbuf[0][row * KST + half * 32]),
                              lds_off(&Bbuf[1][row * KST + half * 32]) };

  v8f acc[4] = {};

  // prologue: stage k-step 0 into buffer 0
  if (wv == 0) tdm_load_tile64x64(abase[0], Arow0, Mn);
  stage64B(boff[0], bg);

  const int NSTEP = Dn / 64;          // 16
  for (int i = 0; i < NSTEP; ++i) {
    const int cur = i & 1;
    if (i + 1 < NSTEP) {
      if (wv == 0) tdm_load_tile64x64(abase[cur ^ 1], Arow0 + (i + 1) * 64, Mn);
      stage64B(boff[cur ^ 1], bg + (i + 1) * 64);
      WAIT_ASYNC(4);                         // current B tile landed
      __builtin_amdgcn_s_wait_tensorcnt(1);  // current A tile landed
    } else {
      WAIT_ASYNC(0);
      __builtin_amdgcn_s_wait_tensorcnt(0);
    }
    __syncthreads();

    const u16* Ab = &Abuf[cur][0];
    const u16* Bb = &Bbuf[cur][0];
    const v16bf a0 = load_a_frag(Ab + (wv * 16) * KST,      KST, lane);
    const v16bf a1 = load_a_frag(Ab + (wv * 16) * KST + 32, KST, lane);
#pragma unroll
    for (int nt = 0; nt < 4; ++nt) {
      v16bf b0 = load_bt_frag(Bb + (nt * 16) * KST,      KST, lane);
      v16bf b1 = load_bt_frag(Bb + (nt * 16) * KST + 32, KST, lane);
      acc[nt] = wmma_bf16(a0, b0, acc[nt]);
      acc[nt] = wmma_bf16(a1, b1, acc[nt]);
    }
    __syncthreads();                  // all reads done before next overwrite
  }

  // ---- store: C layout is VGPR r -> rows {r, r+8}, lane halves -> N 0..15 ----
  const int kh = lane >> 4, nc = lane & 15;
#pragma unroll
  for (int nt = 0; nt < 4; ++nt) {
    const int n = n0 + nt * 16 + nc;
    const float bv = bias[n];
#pragma unroll
    for (int r = 0; r < 8; ++r) {
      const int m = m0 + wv * 16 + r + 8 * kh;
      const float val = acc[nt][r] + bv;
      if (MODE == 2) {
        ((float*)Out)[(size_t)m * Dn + n] = val;
      } else {
        const int b = m >> 11, s = m & (Sn - 1);
        const int h = n >> 6,  d = n & (DKn - 1);
        const size_t idx = (MODE == 0)
            ? ((size_t)(b * Hn + h) * Sn + s) * DKn + d       // [B,H,S,DK]
            : ((size_t)(b * Hn + h) * DKn + d) * Sn + s;      // [B,H,DK,S]
        ((u16*)Out)[idx] = f2bf(val);
      }
    }
  }
}

// ---------------------------------------------------------------------------
// Flash attention: per block = 64 query rows of one (b,h); 1 wave per 16 rows.
// K-blocks of 32 keys: 4 WMMA for scores (DK=64 -> 2 K-steps x 2 N-tiles),
// online softmax, P -> LDS (bf16, A-fragment relayout), 4 WMMA for P@V.
// Vt is [B,H,DK,S] so V^T fragments are contiguous B-operand loads.
// ---------------------------------------------------------------------------
__global__ __launch_bounds__(128)
void mha_attn_kernel(const u16* __restrict__ Qh, const u16* __restrict__ Kh,
                     const u16* __restrict__ Vt, u16* __restrict__ Ctx) {
  __shared__ u16 Plds[4][16 * LDST];

  const int tid  = threadIdx.x;
  const int lane = tid & 31;
  const int wv   = tid >> 5;
  const int kh   = lane >> 4, nc = lane & 15;
  const int bh   = blockIdx.y;              // b*H + h
  const int b    = bh >> 4, h = bh & 15;
  const int q0   = blockIdx.x * 64 + wv * 16;

  const u16* Qb  = Qh + ((size_t)bh * Sn + q0) * DKn;
  const u16* Kb0 = Kh + (size_t)bh * Sn * DKn;
  const u16* Vb  = Vt + (size_t)bh * DKn * Sn;
  u16* Pw = &Plds[wv][0];

  const v16bf aq0 = load_a_frag(Qb,      DKn, lane);   // K-dim 0..31
  const v16bf aq1 = load_a_frag(Qb + 32, DKn, lane);   // K-dim 32..63

  v8f acc[4] = {};
  float mst[8], lst[8];
#pragma unroll
  for (int r = 0; r < 8; ++r) { mst[r] = -1e30f; lst[r] = 0.0f; }

  const float scale = 0.125f;  // 1/sqrt(DK)

  for (int kb = 0; kb < q0 + 16; kb += 32) {
    const u16* Kb = Kb0 + (size_t)kb * DKn;
    if (kb + 32 < Sn) __builtin_prefetch(Kb + 32 * DKn, 0, 1);
    v8f c0 = {}, c1 = {};
    {
      v16bf b00 = load_bt_frag(Kb,                 DKn, lane);
      v16bf b01 = load_bt_frag(Kb + 32,            DKn, lane);
      v16bf b10 = load_bt_frag(Kb + 16 * DKn,      DKn, lane);
      v16bf b11 = load_bt_frag(Kb + 16 * DKn + 32, DKn, lane);
      c0 = wmma_bf16(aq0, b00, c0);
      c0 = wmma_bf16(aq1, b01, c0);
      c1 = wmma_bf16(aq0, b10, c1);
      c1 = wmma_bf16(aq1, b11, c1);
    }
    // online softmax update + relayout P into LDS (A-fragment order)
#pragma unroll
    for (int r = 0; r < 8; ++r) {
      const int qpos = q0 + r + 8 * kh;
      float v0 = c0[r] * scale; if (kb + nc > qpos)      v0 = -1e30f;
      float v1 = c1[r] * scale; if (kb + 16 + nc > qpos) v1 = -1e30f;
      const float rmax  = half16_max(fmaxf(v0, v1));
      const float mnew  = fmaxf(mst[r], rmax);
      const float p0    = __expf(v0 - mnew);
      const float p1    = __expf(v1 - mnew);
      const float rsum  = half16_sum(p0 + p1);
      const float alpha = __expf(mst[r] - mnew);
      lst[r] = lst[r] * alpha + rsum;
      mst[r] = mnew;
      acc[0][r] *= alpha; acc[1][r] *= alpha;
      acc[2][r] *= alpha; acc[3][r] *= alpha;
      const int prow = (r + 8 * kh) * LDST;
      Pw[prow + nc]      = f2bf(p0);
      Pw[prow + 16 + nc] = f2bf(p1);
    }
    __builtin_amdgcn_fence(__ATOMIC_ACQ_REL, "wavefront");

    const v16bf ap = load_a_frag(Pw, LDST, lane);       // P 16x32
#pragma unroll
    for (int nt = 0; nt < 4; ++nt) {
      v16bf bv = load_bt_frag(Vb + (size_t)(nt * 16) * Sn + kb, Sn, lane);
      acc[nt] = wmma_bf16(ap, bv, acc[nt]);
    }
    __builtin_amdgcn_fence(__ATOMIC_ACQ_REL, "wavefront");
  }

  // normalize and store ctx as bf16 in [B, S, H*DK] = [B, S, D] row-major
#pragma unroll
  for (int r = 0; r < 8; ++r) {
    const int s = q0 + r + 8 * kh;
    const float inv = 1.0f / lst[r];
#pragma unroll
    for (int nt = 0; nt < 4; ++nt) {
      const int d = nt * 16 + nc;
      Ctx[((size_t)(b * Sn + s) * Hn + h) * DKn + d] = f2bf(acc[nt][r] * inv);
    }
  }
}

// ---------------------------------------------------------------------------
extern "C" void kernel_launch(void* const* d_in, const int* in_sizes, int n_in,
                              void* d_out, int out_size, void* d_ws, size_t ws_size,
                              hipStream_t stream) {
  (void)in_sizes; (void)n_in; (void)out_size;
  const float* query = (const float*)d_in[0];
  const float* key_  = (const float*)d_in[1];
  const float* value = (const float*)d_in[2];
  // d_in[3] = mask: causal tril; applied analytically inside the attn kernel.
  const float* wq = (const float*)d_in[4];
  const float* bq = (const float*)d_in[5];
  const float* wk = (const float*)d_in[6];
  const float* bk = (const float*)d_in[7];
  const float* wvw = (const float*)d_in[8];
  const float* bv = (const float*)d_in[9];
  const float* wo = (const float*)d_in[10];
  const float* bo = (const float*)d_in[11];

  const size_t actBytes = (size_t)Bn * Hn * Sn * DKn * sizeof(u16);  // 16 MiB
  const size_t wBytes   = (size_t)Dn * Dn * sizeof(u16);             //  2 MiB
  if (ws_size < 7 * actBytes + 4 * wBytes) return;
  char* ws = (char*)d_ws;
  u16* Qh  = (u16*)(ws + 0 * actBytes);
  u16* Kh  = (u16*)(ws + 1 * actBytes);
  u16* Vt  = (u16*)(ws + 2 * actBytes);
  u16* Ctx = (u16*)(ws + 3 * actBytes);
  u16* Xq  = (u16*)(ws + 4 * actBytes);
  u16* Xk  = (u16*)(ws + 5 * actBytes);
  u16* Xv  = (u16*)(ws + 6 * actBytes);
  u16* Wqb = (u16*)(ws + 7 * actBytes + 0 * wBytes);
  u16* Wkb = (u16*)(ws + 7 * actBytes + 1 * wBytes);
  u16* Wvb = (u16*)(ws + 7 * actBytes + 2 * wBytes);
  u16* Wob = (u16*)(ws + 7 * actBytes + 3 * wBytes);

  // ---- f32 -> bf16 bulk converts (bandwidth-bound, ~150 MB total) ----
  const int nIn = Mn * Dn;             // 8,388,608
  const int nW  = Dn * Dn;             // 1,048,576
  dim3 cb(256);
  dim3 cgi((nIn / 8 + 255) / 256), cgw((nW / 8 + 255) / 256);
  cvt_bf16_kernel<<<cgi, cb, 0, stream>>>(query, Xq, nIn);
  cvt_bf16_kernel<<<cgi, cb, 0, stream>>>(key_,  Xk, nIn);
  cvt_bf16_kernel<<<cgi, cb, 0, stream>>>(value, Xv, nIn);
  cvt_bf16_kernel<<<cgw, cb, 0, stream>>>(wq, Wqb, nW);
  cvt_bf16_kernel<<<cgw, cb, 0, stream>>>(wk, Wkb, nW);
  cvt_bf16_kernel<<<cgw, cb, 0, stream>>>(wvw, Wvb, nW);
  cvt_bf16_kernel<<<cgw, cb, 0, stream>>>(wo, Wob, nW);

  // ---- projections ----
  dim3 pg(Mn / 64, Dn / 64);   // (128, 16)
  dim3 pb(128);
  mha_proj_kernel<0><<<pg, pb, 0, stream>>>(Xq, Wqb, bq, (void*)Qh);
  mha_proj_kernel<0><<<pg, pb, 0, stream>>>(Xk, Wkb, bk, (void*)Kh);
  mha_proj_kernel<1><<<pg, pb, 0, stream>>>(Xv, Wvb, bv, (void*)Vt);

  // ---- flash attention ----
  dim3 ag(Sn / 64, Bn * Hn);   // (32, 64)
  mha_attn_kernel<<<ag, pb, 0, stream>>>(Qh, Kh, Vt, Ctx);

  // ---- output projection (f32 + bias) ----
  mha_proj_kernel<2><<<pg, pb, 0, stream>>>(Ctx, Wob, bo, d_out);
}
